// ARCTransformerEncoderLayerWithAttn_33655363731495
// MI455X (gfx1250) — matrix operand order repaired and checked by reading, not compiled, not run on hardware
//
#include <hip/hip_runtime.h>

// ---------------------------------------------------------------------------
// Problem constants (from the reference)
// ---------------------------------------------------------------------------
constexpr int Bn = 8, Sn = 1024, En = 768, Hn = 12, Mn = 3072, Dn = 64;
constexpr int Tn = Bn * Sn;   // 8192 tokens
constexpr int E3 = 3 * En;    // 2304

typedef __bf16 bf16x16 __attribute__((ext_vector_type(16)));
typedef float  f32x8   __attribute__((ext_vector_type(8)));

union FragAB { bf16x16 v; unsigned int u[8]; };
union FragC  { f32x8  v; float        f[8]; };
union U4S    { uint4  q; unsigned short s[8]; };

static __device__ __forceinline__ unsigned short f2bf(float x) {
  unsigned int u = __float_as_uint(x);
  u += 0x7fffu + ((u >> 16) & 1u);          // round-to-nearest-even
  return (unsigned short)(u >> 16);
}

// ---------------------------------------------------------------------------
// CDNA5 async global->LDS copy (ASYNCcnt path), per CDNA5 ISA
// VDST VGPR carries the LDS byte address; generic-pointer low 32 bits are the
// LDS offset (aperture lives in the high half).
// ---------------------------------------------------------------------------
static __device__ __forceinline__ void async_ld_b128(void* lds_dst,
                                                     const void* gsrc) {
  unsigned l = (unsigned)(unsigned long long)lds_dst;
  asm volatile("global_load_async_to_lds_b128 %0, %1, off"
               :: "v"(l), "v"(gsrc)
               : "memory");
}
static __device__ __forceinline__ void wait_async0() {
  asm volatile("s_wait_asynccnt 0x0" ::: "memory");
}

// ---------------------------------------------------------------------------
// fp32 -> bf16 conversion
// ---------------------------------------------------------------------------
__global__ void k_cvt_bf16(const float* __restrict__ in,
                           unsigned short* __restrict__ out, int n) {
  int i = blockIdx.x * blockDim.x + threadIdx.x;
  if (i < n) out[i] = f2bf(in[i]);
}

// ---------------------------------------------------------------------------
// Generic WMMA GEMM:  C[M,N] = A[M,K] @ W[N,K]^T + bias   (A,W bf16, acc f32)
// Block tile 128x64, BK=32, 256 threads = 8 waves, wave w -> rows w*16..+15.
// Double-buffered LDS fed by async global->LDS DMA; compute overlaps copy.
// ---------------------------------------------------------------------------
template <bool GELU, bool OUTBF16>
__global__ __launch_bounds__(256)
void k_gemm(const unsigned short* __restrict__ A,
            const unsigned short* __restrict__ W,
            const float* __restrict__ bias,
            float* __restrict__ Cf, unsigned short* __restrict__ Cb,
            int N, int K) {
  __shared__ __align__(16) unsigned int As[2][128 * 16];  // [buf][row][k-pair]
  __shared__ __align__(16) unsigned int Bs[2][64 * 16];   // [buf][n]  [k-pair]
  const int tid  = threadIdx.x;
  const int lane = tid & 31, w = tid >> 5;
  const int hi   = lane >> 4, ln = lane & 15;
  const int row0 = blockIdx.y * 128;
  const int col0 = blockIdx.x * 64;
  const int Ku4  = K >> 3;                                // uint4 per row
  const int nK   = K / 32;
  const uint4* A4 = (const uint4*)A;
  const uint4* W4 = (const uint4*)W;

  auto stage = [&](int kt) {
    const int buf = kt & 1;
    uint4* As4 = (uint4*)As[buf];
    uint4* Bs4 = (uint4*)Bs[buf];
#pragma unroll
    for (int rep = 0; rep < 2; ++rep) {
      int j = tid + rep * 256;
      async_ld_b128(&As4[j],
                    &A4[(size_t)(row0 + (j >> 2)) * Ku4 + kt * 4 + (j & 3)]);
    }
    async_ld_b128(&Bs4[tid],
                  &W4[(size_t)(col0 + (tid >> 2)) * Ku4 + kt * 4 + (tid & 3)]);
  };

  FragC acc[4];
#pragma unroll
  for (int n = 0; n < 4; ++n)
#pragma unroll
    for (int r = 0; r < 8; ++r) acc[n].f[r] = 0.0f;

  stage(0);
  for (int kt = 0; kt < nK; ++kt) {
    wait_async0();          // own DMA for tile kt landed
    __syncthreads();        // everyone's landed; everyone done with buf kt-1
    if (kt + 1 < nK) stage(kt + 1);   // overlap next DMA with this compute

    const unsigned int* Ab = As[kt & 1];
    const unsigned int* Bb = Bs[kt & 1];
    FragAB a;
    const int arowL = (w << 4) + ln;
#pragma unroll
    for (int i = 0; i < 8; ++i)
      a.u[i] = Ab[arowL * 16 + ((i & 3) + ((i >> 2) << 3) + (hi << 2))];
#pragma unroll
    for (int n = 0; n < 4; ++n) {
      FragAB b;
#pragma unroll
      for (int i = 0; i < 8; ++i)
        b.u[i] = Bb[((n << 4) + ln) * 16 + ((i & 3) + ((i >> 2) << 3) + (hi << 2))];
      acc[n].v = __builtin_amdgcn_wmma_f32_16x16x32_bf16(
          false, a.v, false, b.v, (short)0, acc[n].v, false, false);
    }
  }

  // ---- epilogue: bias (+GELU), f32 or bf16 store ----
#pragma unroll
  for (int n = 0; n < 4; ++n) {
    int gcol = col0 + (n << 4) + ln;
    float bv = bias[gcol];
#pragma unroll
    for (int r = 0; r < 8; ++r) {
      int grow = row0 + (w << 4) + r + (hi << 3);
      float v = acc[n].f[r] + bv;
      if (GELU) v = 0.5f * v * (1.0f + erff(v * 0.70710678118654752f));
      size_t idx = (size_t)grow * N + gcol;
      if (OUTBF16) Cb[idx] = f2bf(v);
      else         Cf[idx] = v;
    }
  }
}

// ---------------------------------------------------------------------------
// RoPE + split heads: qkv[T,3E] f32 -> q/k/v [B,H,S,D] bf16 (q,k rotated)
// ---------------------------------------------------------------------------
__global__ __launch_bounds__(256)
void k_rope_split(const float* __restrict__ qkv,
                  const float* __restrict__ cosT, const float* __restrict__ sinT,
                  unsigned short* __restrict__ qh, unsigned short* __restrict__ kh,
                  unsigned short* __restrict__ vh) {
  const int token = blockIdx.x;          // b*S + s
  const int b = token / Sn, s = token - b * Sn;
  const float* row = qkv + (size_t)token * E3;
  for (int i = threadIdx.x; i < E3; i += blockDim.x) {
    int part = i / En, e = i - part * En;
    int h = e / Dn, d = e - h * Dn;
    float v = row[i], o;
    if (part == 2) {
      o = v;
    } else {
      float c = cosT[s * Dn + d], sn = sinT[s * Dn + d];
      float partner = (d & 1) ? row[i - 1] : row[i + 1];
      o = (d & 1) ? (v * c + partner * sn) : (v * c - partner * sn);
    }
    size_t dst = ((size_t)(b * Hn + h) * Sn + s) * Dn + d;
    (part == 0 ? qh : part == 1 ? kh : vh)[dst] = f2bf(o);
  }
}

// ---------------------------------------------------------------------------
// Flash attention: one block per (b*H, 128 query rows); 8 waves x 16 q-rows.
// Online softmax; WMMA for Q@K^T and P@V. K tile staged via async DMA,
// overlapped with the V transpose. ctx written bf16 as [token][h*D+d].
// ---------------------------------------------------------------------------
__global__ __launch_bounds__(256)
void k_attn(const unsigned short* __restrict__ Q,
            const unsigned short* __restrict__ Kh,
            const unsigned short* __restrict__ V,
            const unsigned char* __restrict__ mask,
            unsigned short* __restrict__ ctx) {
  __shared__ __align__(16) unsigned int  Ks[32 * 32];        // [key][d-pair]
  __shared__ __align__(16) unsigned int  Vs[64 * 16];        // [d][key-pair]
  __shared__ __align__(16) unsigned short Ps[8 * 16 * 32];   // per-wave P tile
  __shared__ float ms[32];

  const int tid  = threadIdx.x;
  const int lane = tid & 31, w = tid >> 5;
  const int hi   = lane >> 4, ln = lane & 15;
  const int bh = blockIdx.y, b = bh / Hn, h = bh - b * Hn;
  const size_t rowbase = (size_t)bh * Sn;
  const int q0 = blockIdx.x * 128 + w * 16;

  const unsigned int* Qu = (const unsigned int*)Q;
  const uint4* K4 = (const uint4*)Kh;
  const uint4* V4 = (const uint4*)V;
  uint4* Ks4 = (uint4*)Ks;
  unsigned short* Vs16 = (unsigned short*)Vs;
  unsigned int*   Psu  = (unsigned int*)Ps;

  // ---- Q fragments stay in registers for the whole key loop ----
  FragAB qa[2];
  {
    size_t qrow = rowbase + q0 + ln;     // [bh][s] row, 32 uints (=64 bf16)
#pragma unroll
    for (int f = 0; f < 2; ++f)
#pragma unroll
      for (int i = 0; i < 8; ++i)
        qa[f].u[i] = Qu[qrow * 32 + f * 16 +
                        ((i & 3) + ((i >> 2) << 3) + (hi << 2))];
  }

  FragC o[4];
  float mrow[8], lrow[8];
#pragma unroll
  for (int n = 0; n < 4; ++n)
#pragma unroll
    for (int r = 0; r < 8; ++r) o[n].f[r] = 0.0f;
#pragma unroll
  for (int r = 0; r < 8; ++r) { mrow[r] = -1e30f; lrow[r] = 0.0f; }

  for (int kt = 0; kt < Sn / 32; ++kt) {
    const int key0 = kt * 32;
    // ---- stage K tile via async DMA; V transposed via VGPRs; mask bias ----
    {
      int key = tid >> 3, c = tid & 7;
      async_ld_b128(&Ks4[tid], &K4[(rowbase + key0 + key) * 8 + c]);
      int idx = tid * 8, vkey = idx >> 6, d0 = idx & 63;
      U4S uu; uu.q = V4[(rowbase + key0 + vkey) * 8 + (d0 >> 3)];
#pragma unroll
      for (int jj = 0; jj < 8; ++jj) Vs16[(d0 + jj) * 32 + vkey] = uu.s[jj];
      if (tid < 32)
        ms[tid] = mask[(size_t)b * Sn + key0 + tid] ? -1e30f : 0.0f;
    }
    wait_async0();
    __syncthreads();

    // ---- scores = Q @ K^T  (two 16x16 key sub-tiles, K-dim = D = 64) ----
    FragC sf[2];
#pragma unroll
    for (int ks = 0; ks < 2; ++ks) {
#pragma unroll
      for (int r = 0; r < 8; ++r) sf[ks].f[r] = 0.0f;
#pragma unroll
      for (int f = 0; f < 2; ++f) {
        FragAB bk;
#pragma unroll
        for (int i = 0; i < 8; ++i)
          bk.u[i] = Ks[((ks << 4) + ln) * 32 + (f << 4) + ((i & 3) + ((i >> 2) << 3) + (hi << 2))];
        sf[ks].v = __builtin_amdgcn_wmma_f32_16x16x32_bf16(
            false, qa[f].v, false, bk.v, (short)0, sf[ks].v, false, false);
      }
    }

    // ---- online softmax per row (16 lanes of a half own one row) ----
    const float scale = 0.125f;  // D^-0.5
#pragma unroll
    for (int r = 0; r < 8; ++r) {
      float s0 = sf[0].f[r] * scale + ms[ln];
      float s1 = sf[1].f[r] * scale + ms[16 + ln];
      float tmax = fmaxf(s0, s1);
#pragma unroll
      for (int mmask = 1; mmask < 16; mmask <<= 1)
        tmax = fmaxf(tmax, __shfl_xor(tmax, mmask, 32));
      float mn = fmaxf(mrow[r], tmax);
      float alpha = __expf(mrow[r] - mn);
      float p0 = __expf(s0 - mn), p1 = __expf(s1 - mn);
      float ts = p0 + p1;
#pragma unroll
      for (int mmask = 1; mmask < 16; mmask <<= 1)
        ts += __shfl_xor(ts, mmask, 32);
      lrow[r] = lrow[r] * alpha + ts;
      mrow[r] = mn;
#pragma unroll
      for (int n = 0; n < 4; ++n) o[n].f[r] *= alpha;
      int m = r + (hi << 3);
      Ps[w * 512 + m * 32 + ln]      = f2bf(p0);
      Ps[w * 512 + m * 32 + 16 + ln] = f2bf(p1);
    }
    __syncthreads();

    // ---- ctx += P @ V  (A-frag from LDS transpose of C-layout P) ----
    FragAB pa;
#pragma unroll
    for (int i = 0; i < 8; ++i)
      pa.u[i] = Psu[w * 256 + ln * 16 +
                    ((i & 3) + ((i >> 2) << 3) + (hi << 2))];
#pragma unroll
    for (int n = 0; n < 4; ++n) {
      FragAB bv;
#pragma unroll
      for (int i = 0; i < 8; ++i)
        bv.u[i] = Vs[((n << 4) + ln) * 16 + ((i & 3) + ((i >> 2) << 3) + (hi << 2))];
      o[n].v = __builtin_amdgcn_wmma_f32_16x16x32_bf16(
          false, pa.v, false, bv.v, (short)0, o[n].v, false, false);
    }
    __syncthreads();
  }

  // ---- normalize + store ctx as [token][h*D + d] bf16 ----
#pragma unroll
  for (int r = 0; r < 8; ++r) {
    float inv = 1.0f / lrow[r];
    int srow = q0 + r + (hi << 3);
    size_t base = ((size_t)(b * Sn + srow)) * En + h * Dn;
#pragma unroll
    for (int n = 0; n < 4; ++n)
      ctx[base + (n << 4) + ln] = f2bf(o[n].f[r] * inv);
  }
}

// ---------------------------------------------------------------------------
// Fused residual add + LayerNorm; optional f32 / bf16 outputs.
// One block (256 thr) per token row of 768.
// ---------------------------------------------------------------------------
__global__ __launch_bounds__(256)
void k_add_ln(const float* __restrict__ xa, const float* __restrict__ xb,
              const float* __restrict__ g, const float* __restrict__ be,
              float* __restrict__ outf, unsigned short* __restrict__ outb) {
  __shared__ float red[256];
  const size_t base = (size_t)blockIdx.x * En;
  float v[3], s = 0.0f;
#pragma unroll
  for (int j = 0; j < 3; ++j) {
    int c = threadIdx.x + j * 256;
    v[j] = xa[base + c] + xb[base + c];
    s += v[j];
  }
  red[threadIdx.x] = s; __syncthreads();
  for (int st = 128; st > 0; st >>= 1) {
    if (threadIdx.x < st) red[threadIdx.x] += red[threadIdx.x + st];
    __syncthreads();
  }
  float mu = red[0] * (1.0f / En);
  __syncthreads();
  float sq = 0.0f;
#pragma unroll
  for (int j = 0; j < 3; ++j) { float d = v[j] - mu; sq += d * d; }
  red[threadIdx.x] = sq; __syncthreads();
  for (int st = 128; st > 0; st >>= 1) {
    if (threadIdx.x < st) red[threadIdx.x] += red[threadIdx.x + st];
    __syncthreads();
  }
  float rstd = rsqrtf(red[0] * (1.0f / En) + 1e-5f);
#pragma unroll
  for (int j = 0; j < 3; ++j) {
    int c = threadIdx.x + j * 256;
    float y = (v[j] - mu) * rstd * g[c] + be[c];
    if (outf) outf[base + c] = y;
    if (outb) outb[base + c] = f2bf(y);
  }
}

// ---------------------------------------------------------------------------
// Host launcher
// ---------------------------------------------------------------------------
extern "C" void kernel_launch(void* const* d_in, const int* in_sizes, int n_in,
                              void* d_out, int out_size, void* d_ws, size_t ws_size,
                              hipStream_t stream) {
  (void)in_sizes; (void)n_in; (void)out_size; (void)ws_size;
  const float* x        = (const float*)d_in[0];
  const unsigned char* kpm = (const unsigned char*)d_in[1];
  const float* qkv_w    = (const float*)d_in[2];
  const float* qkv_b    = (const float*)d_in[3];
  const float* proj_w   = (const float*)d_in[4];
  const float* proj_b   = (const float*)d_in[5];
  const float* ln1_g    = (const float*)d_in[6];
  const float* ln1_b    = (const float*)d_in[7];
  const float* w1       = (const float*)d_in[8];
  const float* b1       = (const float*)d_in[9];
  const float* w2       = (const float*)d_in[10];
  const float* b2       = (const float*)d_in[11];
  const float* ln2_g    = (const float*)d_in[12];
  const float* ln2_b    = (const float*)d_in[13];
  const float* rope_cos = (const float*)d_in[14];
  const float* rope_sin = (const float*)d_in[15];

  char* base = (char*)d_ws;
  size_t off = 0;
  auto take = [&](size_t bytes) -> char* {
    size_t o = (off + 255) & ~(size_t)255;
    off = o + bytes;
    return base + o;
  };
  unsigned short* xb    = (unsigned short*)take((size_t)Tn * En * 2);
  unsigned short* wqkvb = (unsigned short*)take((size_t)E3 * En * 2);
  unsigned short* wprjb = (unsigned short*)take((size_t)En * En * 2);
  unsigned short* w1b   = (unsigned short*)take((size_t)Mn * En * 2);
  unsigned short* w2b   = (unsigned short*)take((size_t)En * Mn * 2);
  float*          qkvf  = (float*)take((size_t)Tn * E3 * 4);
  unsigned short* qh    = (unsigned short*)take((size_t)Tn * En * 2);
  unsigned short* kh    = (unsigned short*)take((size_t)Tn * En * 2);
  unsigned short* vh    = (unsigned short*)take((size_t)Tn * En * 2);
  unsigned short* ctxb  = (unsigned short*)take((size_t)Tn * En * 2);
  float*          attnf = (float*)take((size_t)Tn * En * 4);
  float*          y1f   = (float*)take((size_t)Tn * En * 4);
  unsigned short* y1b   = (unsigned short*)take((size_t)Tn * En * 2);
  unsigned short* hb    = (unsigned short*)take((size_t)Tn * Mn * 2);
  float*          ffnf  = (float*)take((size_t)Tn * En * 4);

  auto cvt = [&](const float* src, unsigned short* dst, size_t n) {
    k_cvt_bf16<<<dim3((unsigned)((n + 255) / 256)), 256, 0, stream>>>(src, dst, (int)n);
  };
  cvt(x,      xb,    (size_t)Tn * En);
  cvt(qkv_w,  wqkvb, (size_t)E3 * En);
  cvt(proj_w, wprjb, (size_t)En * En);
  cvt(w1,     w1b,   (size_t)Mn * En);
  cvt(w2,     w2b,   (size_t)En * Mn);

  // QKV projection: [T,E] @ [3E,E]^T
  k_gemm<false, false><<<dim3(E3 / 64, Tn / 128), 256, 0, stream>>>(
      xb, wqkvb, qkv_b, qkvf, nullptr, E3, En);

  // RoPE + head split
  k_rope_split<<<dim3(Tn), 256, 0, stream>>>(qkvf, rope_cos, rope_sin, qh, kh, vh);

  // Flash attention
  k_attn<<<dim3(Sn / 128, Bn * Hn), 256, 0, stream>>>(qh, kh, vh, kpm, ctxb);

  // Output projection
  k_gemm<false, false><<<dim3(En / 64, Tn / 128), 256, 0, stream>>>(
      ctxb, wprjb, proj_b, attnf, nullptr, En, En);

  // Residual + LN1 (f32 for residual chain, bf16 for FC1)
  k_add_ln<<<dim3(Tn), 256, 0, stream>>>(x, attnf, ln1_g, ln1_b, y1f, y1b);

  // FC1 + exact GELU -> bf16
  k_gemm<true, true><<<dim3(Mn / 64, Tn / 128), 256, 0, stream>>>(
      y1b, w1b, b1, nullptr, hb, Mn, En);

  // FC2
  k_gemm<false, false><<<dim3(En / 64, Tn / 128), 256, 0, stream>>>(
      hb, w2b, b2, ffnf, nullptr, En, Mn);

  // Residual + LN2 -> d_out (f32)
  k_add_ln<<<dim3(Tn), 256, 0, stream>>>(y1f, ffnf, ln2_g, ln2_b,
                                         (float*)d_out, nullptr);
}
